// AttentionCoder_76965813944640
// MI455X (gfx1250) — matrix operand order repaired and checked by reading, not compiled, not run on hardware
//
#include <hip/hip_runtime.h>
#include <hip/hip_bf16.h>

typedef __attribute__((ext_vector_type(16))) __bf16 v16bf;
typedef __attribute__((ext_vector_type(8)))  __bf16 v8bf;
typedef __attribute__((ext_vector_type(8)))  float  v8f;
typedef __attribute__((ext_vector_type(4)))  float  v4f;

#define BATCH 4
#define SEQ   1024
#define DIM   1024
#define NH    16
#define HDIM  64

static __device__ __forceinline__ __bf16 f2bf(float f) {
  unsigned u = __builtin_bit_cast(unsigned, f);
  u += 0x7fffu + ((u >> 16) & 1u);            // round to nearest even
  unsigned short h = (unsigned short)(u >> 16);
  return __builtin_bit_cast(__bf16, h);
}

// Async global->LDS copy of 16 bytes (per lane). LDS offset = low 32 bits of
// the flat shared-pointer address.
static __device__ __forceinline__ void async_cp16(void* lds_dst, const void* gsrc) {
  unsigned lds = (unsigned)(size_t)lds_dst;
  unsigned long long ga = (unsigned long long)(size_t)gsrc;
  asm volatile("global_load_async_to_lds_b128 %0, %1, off"
               :: "v"(lds), "v"(ga) : "memory");
}
static __device__ __forceinline__ void async_wait0() {
  asm volatile("s_wait_asynccnt 0x0" ::: "memory");
}

// A-matrix 16x32 bf16 fragment: lane holds row m = lane%16 (hl = lane/16).
// element j<8  -> K = kc + 8*hl + j ; element j>=8 -> K = kc + 16 + 8*hl + (j-8)
static __device__ __forceinline__ v16bf ldfragA(const __bf16* base, int pitch,
                                                int m, int kc, int hl) {
  const __bf16* p = base + (size_t)m * pitch + kc + hl * 8;
  v8bf lo = *(const v8bf*)p;
  v8bf hi = *(const v8bf*)(p + 16);
  v16bf a;
#pragma unroll
  for (int j = 0; j < 8; ++j) { a[j] = lo[j]; a[j + 8] = hi[j]; }
  return a;
}

// B-matrix 32x16 bf16 fragment from K-major (transposed) storage:
// lane holds column n = lane%16; element j<8 -> K = kc + 8*hl + j ; j>=8 -> K = kc + 16 + 8*hl + (j-8)
static __device__ __forceinline__ v16bf ldfragB(const __bf16* base, int pitch,
                                                int n, int kc, int hl) {
  const __bf16* p = base + (size_t)n * pitch + kc + hl * 8;
  v8bf lo = *(const v8bf*)p;
  v8bf hi = *(const v8bf*)(p + 16);
  v16bf b;
#pragma unroll
  for (int j = 0; j < 8; ++j) { b[j] = lo[j]; b[j + 8] = hi[j]; }
  return b;
}

__global__ void cvt_f32_bf16(const float* __restrict__ s, __bf16* __restrict__ d, int n) {
  int i = (blockIdx.x * blockDim.x + threadIdx.x) * 8;
  if (i < n) {
    v4f a = *(const v4f*)(s + i);
    v4f b = *(const v4f*)(s + i + 4);
    v8bf o;
#pragma unroll
    for (int j = 0; j < 4; ++j) { o[j] = f2bf(a[j]); o[j + 4] = f2bf(b[j]); }
    *(v8bf*)(d + i) = o;
  }
}

// C[M,N] = A[M,K] @ B[K,N].  A,B bf16 row-major, f32 accumulate.
// Tile 128x128, K-step 64, 8 waves x (2x4) 16x16 accum subtiles.
// A tile staged via async global->LDS; B tile register-transposed into LDS.
template <bool OUT_BF16>
__global__ void __launch_bounds__(256) gemm_bf16(const __bf16* __restrict__ A,
                                                 const __bf16* __restrict__ B,
                                                 float* __restrict__ Cf,
                                                 __bf16* __restrict__ Cb,
                                                 int N, int K) {
  __shared__ __bf16 As[128][72];   // [m][k], padded pitch
  __shared__ __bf16 Bs[128][72];   // [n][k], K-major (transposed on fill)

  const int tid  = threadIdx.x;
  const int lane = tid & 31;
  const int l16  = lane & 15;
  const int hl   = lane >> 4;
  const int w    = tid >> 5;            // 8 waves
  const int wm0  = (w & 3) * 32;        // wave sub-tile: 32 rows x 64 cols
  const int wn0  = (w >> 2) * 64;
  const int m0   = blockIdx.y * 128;
  const int n0   = blockIdx.x * 128;

  v8f c[2][4];
#pragma unroll
  for (int i = 0; i < 2; ++i)
#pragma unroll
    for (int j = 0; j < 4; ++j)
#pragma unroll
      for (int e = 0; e < 8; ++e) c[i][j][e] = 0.0f;

  const int nk = K >> 6;
  for (int kk = 0; kk < nk; ++kk) {
    const int k0 = kk << 6;
    __syncthreads();
    // stage A tile 128x64: async global->LDS (ASYNCcnt path, no VGPR bounce)
#pragma unroll
    for (int i = 0; i < 4; ++i) {
      int lin = tid + (i << 8);
      int row = lin >> 3, c8 = (lin & 7) << 3;
      async_cp16(&As[row][c8], &A[(size_t)(m0 + row) * K + k0 + c8]);
    }
    // stage B tile 64x128 transposed into [n][k]; pack K-pairs as b32 stores
#pragma unroll
    for (int i = 0; i < 2; ++i) {
      int lin = tid + (i << 8);
      int kr2 = (lin >> 4) << 1, n8 = (lin & 15) << 3;
      v8bf va = *(const v8bf*)&B[(size_t)(k0 + kr2) * N + n0 + n8];
      v8bf vb = *(const v8bf*)&B[(size_t)(k0 + kr2 + 1) * N + n0 + n8];
#pragma unroll
      for (int j = 0; j < 8; ++j) {
        unsigned p = (unsigned)__builtin_bit_cast(unsigned short, va[j]) |
                     ((unsigned)__builtin_bit_cast(unsigned short, vb[j]) << 16);
        *(unsigned*)&Bs[n8 + j][kr2] = p;
      }
    }
    // prefetch next K-step while this one computes (global_prefetch_b8)
    if (kk + 1 < nk) {
      __builtin_prefetch(&A[(size_t)(m0 + (tid >> 3)) * K + k0 + 64 + ((tid & 7) << 3)], 0, 1);
      __builtin_prefetch(&B[(size_t)(k0 + 64 + (tid >> 4)) * N + n0 + ((tid & 15) << 3)], 0, 1);
    }
    async_wait0();
    __syncthreads();

#pragma unroll
    for (int kc = 0; kc < 64; kc += 32) {
      v16bf af[2], bf[4];
#pragma unroll
      for (int sm = 0; sm < 2; ++sm)
        af[sm] = ldfragA(&As[0][0], 72, wm0 + sm * 16 + l16, kc, hl);
#pragma unroll
      for (int sn = 0; sn < 4; ++sn)
        bf[sn] = ldfragB(&Bs[0][0], 72, wn0 + sn * 16 + l16, kc, hl);
#pragma unroll
      for (int sm = 0; sm < 2; ++sm)
#pragma unroll
        for (int sn = 0; sn < 4; ++sn)
          c[sm][sn] = __builtin_amdgcn_wmma_f32_16x16x32_bf16(
              false, af[sm], false, bf[sn], (short)0, c[sm][sn], false, false);
    }
  }

  // epilogue: C/D layout -> row = r + 8*hl, col = lane%16 (compile-time type)
#pragma unroll
  for (int sm = 0; sm < 2; ++sm)
#pragma unroll
    for (int sn = 0; sn < 4; ++sn)
#pragma unroll
      for (int r = 0; r < 8; ++r) {
        size_t row = (size_t)(m0 + wm0 + sm * 16 + r + 8 * hl);
        size_t col = (size_t)(n0 + wn0 + sn * 16 + l16);
        if (OUT_BF16) Cb[row * N + col] = f2bf(c[sm][sn][r]);
        else          Cf[row * N + col] = c[sm][sn][r];
      }
}

// Flash attention: one block per (b, h, 64-query tile); 4 waves x 16 rows.
__global__ void __launch_bounds__(128) attn_fwd(const __bf16* __restrict__ qkv, // [B*L][3*D] bf16
                                                const float* __restrict__ bias, // [H][L][L] f32
                                                __bf16* __restrict__ attn_out)  // [B*L][D] bf16
{
  __shared__ __bf16 Qs[64][72];       // [q][hd]
  __shared__ __bf16 Ks[64][72];       // [key][hd]  (== B^T for Q@K^T)
  __shared__ __bf16 Vt[64][72];       // [hd][key]  (== B^T for P@V)
  __shared__ __bf16 Ps[4][16][72];    // per-wave P staging

  const int tid  = threadIdx.x;
  const int lane = tid & 31;
  const int l16  = lane & 15;
  const int hl   = lane >> 4;
  const int wq   = tid >> 5;                 // wave id: query rows wq*16..+15

  const int qt = blockIdx.x & 15;
  const int h  = (blockIdx.x >> 4) & 15;
  const int bb = blockIdx.x >> 8;
  const int q0 = qt * 64;
  const size_t rowbase = (size_t)bb * SEQ;
  const float scale = 0.125f;                // 1/sqrt(64)

  // load Q tile (64 x 64) via async global->LDS
#pragma unroll
  for (int i = 0; i < 4; ++i) {
    int lin = tid + (i << 7);
    int row = lin >> 3, c8 = (lin & 7) << 3;
    async_cp16(&Qs[row][c8],
               &qkv[(rowbase + q0 + row) * (3 * DIM) + h * HDIM + c8]);
  }

  float m_s[8], l_s[8];
  v8f o[4];
#pragma unroll
  for (int r = 0; r < 8; ++r) { m_s[r] = -__builtin_inff(); l_s[r] = 0.0f; }
#pragma unroll
  for (int st = 0; st < 4; ++st)
#pragma unroll
    for (int e = 0; e < 8; ++e) o[st][e] = 0.0f;

  for (int kt = 0; kt <= qt; ++kt) {
    const int k0 = kt * 64;
    __syncthreads();
    // stage K tile async (row-major); V tile register-transposed
#pragma unroll
    for (int i = 0; i < 4; ++i) {
      int lin = tid + (i << 7);
      int row = lin >> 3, c8 = (lin & 7) << 3;
      async_cp16(&Ks[row][c8],
                 &qkv[(rowbase + k0 + row) * (3 * DIM) + DIM + h * HDIM + c8]);
      v8bf v = *(const v8bf*)&qkv[(rowbase + k0 + row) * (3 * DIM) + 2 * DIM + h * HDIM + c8];
#pragma unroll
      for (int j = 0; j < 8; ++j) Vt[c8 + j][row] = v[j];
    }
    // prefetch next K/V tile
    if (kt < qt) {
      int row = tid >> 3, c8 = (tid & 7) << 3;
      __builtin_prefetch(&qkv[(rowbase + k0 + 64 + row) * (3 * DIM) + DIM + h * HDIM + c8], 0, 1);
      __builtin_prefetch(&qkv[(rowbase + k0 + 64 + row) * (3 * DIM) + 2 * DIM + h * HDIM + c8], 0, 1);
    }
    async_wait0();
    __syncthreads();

    // S = Q @ K^T  (16 rows x 64 keys per wave)
    v8f s[4];
#pragma unroll
    for (int sn = 0; sn < 4; ++sn)
#pragma unroll
      for (int e = 0; e < 8; ++e) s[sn][e] = 0.0f;
#pragma unroll
    for (int kc = 0; kc < 64; kc += 32) {
      v16bf a = ldfragA(&Qs[0][0], 72, wq * 16 + l16, kc, hl);
#pragma unroll
      for (int sn = 0; sn < 4; ++sn) {
        v16bf b = ldfragB(&Ks[0][0], 72, sn * 16 + l16, kc, hl);
        s[sn] = __builtin_amdgcn_wmma_f32_16x16x32_bf16(
            false, a, false, b, (short)0, s[sn], false, false);
      }
    }

    // scale + bias + causal mask
    const int diag = (kt == qt);
#pragma unroll
    for (int sn = 0; sn < 4; ++sn) {
      const int kg = k0 + sn * 16 + l16;
      const float* bp = bias + ((size_t)h * SEQ + (q0 + wq * 16 + hl * 8)) * SEQ + kg;
#pragma unroll
      for (int r = 0; r < 8; ++r) {
        float v = s[sn][r] * scale + bp[(size_t)r * SEQ];
        if (diag && kg > q0 + wq * 16 + r + 8 * hl) v = -__builtin_inff();
        s[sn][r] = v;
      }
    }

    // online softmax (row = 16 lanes of same half-wave)
#pragma unroll
    for (int r = 0; r < 8; ++r) {
      float mx = fmaxf(fmaxf(s[0][r], s[1][r]), fmaxf(s[2][r], s[3][r]));
      mx = fmaxf(mx, __shfl_xor(mx, 1, 32));
      mx = fmaxf(mx, __shfl_xor(mx, 2, 32));
      mx = fmaxf(mx, __shfl_xor(mx, 4, 32));
      mx = fmaxf(mx, __shfl_xor(mx, 8, 32));
      float mnew = fmaxf(m_s[r], mx);
      float corr = __expf(m_s[r] - mnew);
      float ls = 0.0f;
#pragma unroll
      for (int sn = 0; sn < 4; ++sn) {
        float p = __expf(s[sn][r] - mnew);
        s[sn][r] = p;
        ls += p;
      }
      ls += __shfl_xor(ls, 1, 32);
      ls += __shfl_xor(ls, 2, 32);
      ls += __shfl_xor(ls, 4, 32);
      ls += __shfl_xor(ls, 8, 32);
      l_s[r] = l_s[r] * corr + ls;
      m_s[r] = mnew;
#pragma unroll
      for (int st = 0; st < 4; ++st) o[st][r] *= corr;
    }

    // reshape P (C-layout) -> A-layout via per-wave LDS staging (DS in-order per wave)
#pragma unroll
    for (int sn = 0; sn < 4; ++sn)
#pragma unroll
      for (int r = 0; r < 8; ++r)
        Ps[wq][r + 8 * hl][sn * 16 + l16] = f2bf(s[sn][r]);

    // O += P @ V
#pragma unroll
    for (int kc = 0; kc < 64; kc += 32) {
      v16bf a = ldfragA(&Ps[wq][0][0], 72, l16, kc, hl);
#pragma unroll
      for (int st = 0; st < 4; ++st) {
        v16bf b = ldfragB(&Vt[0][0], 72, st * 16 + l16, kc, hl);
        o[st] = __builtin_amdgcn_wmma_f32_16x16x32_bf16(
            false, a, false, b, (short)0, o[st], false, false);
      }
    }
  }

  // normalize and store bf16 [b*L + q][h*64 + hd]
#pragma unroll
  for (int st = 0; st < 4; ++st)
#pragma unroll
    for (int r = 0; r < 8; ++r) {
      float val = o[st][r] * (1.0f / l_s[r]);
      size_t row = rowbase + q0 + wq * 16 + r + 8 * hl;
      attn_out[row * DIM + h * HDIM + st * 16 + l16] = f2bf(val);
    }
}

extern "C" void kernel_launch(void* const* d_in, const int* in_sizes, int n_in,
                              void* d_out, int out_size, void* d_ws, size_t ws_size,
                              hipStream_t stream) {
  const float* x      = (const float*)d_in[0];
  // d_in[1] = mask (causal, known analytically) — unused
  const float* w_qkv  = (const float*)d_in[2];
  const float* w_out  = (const float*)d_in[3];
  const float* bias   = (const float*)d_in[4];

  const size_t nX  = (size_t)BATCH * SEQ * DIM;     // 4,194,304
  const size_t nWq = (size_t)DIM * 3 * DIM;         // 3,145,728
  const size_t nWo = (size_t)DIM * DIM;             // 1,048,576

  __bf16* xb   = (__bf16*)d_ws;
  __bf16* wqb  = xb  + nX;
  __bf16* wob  = wqb + nWq;
  __bf16* qkvb = wob + nWo;                            // [B*L][3*D]
  __bf16* attb = qkvb + (size_t)BATCH * SEQ * 3 * DIM; // [B*L][D]

  cvt_f32_bf16<<<(unsigned)(nX  / 2048), 256, 0, stream>>>(x,     xb,  (int)nX);
  cvt_f32_bf16<<<(unsigned)(nWq / 2048), 256, 0, stream>>>(w_qkv, wqb, (int)nWq);
  cvt_f32_bf16<<<(unsigned)(nWo / 2048), 256, 0, stream>>>(w_out, wob, (int)nWo);

  // qkv = x @ w_qkv  : M=4096, N=3072, K=1024 -> bf16
  gemm_bf16<true><<<dim3(3 * DIM / 128, BATCH * SEQ / 128), 256, 0, stream>>>(
      xb, wqb, nullptr, qkvb, 3 * DIM, DIM);

  // flash attention: B*H*(L/64) blocks
  attn_fwd<<<dim3(BATCH * NH * (SEQ / 64)), 128, 0, stream>>>(qkvb, bias, attb);

  // out = attn @ w_out : M=4096, N=1024, K=1024 -> f32 d_out
  gemm_bf16<false><<<dim3(DIM / 128, BATCH * SEQ / 128), 256, 0, stream>>>(
      attb, wob, (float*)d_out, nullptr, DIM, DIM);
}
